// CrossAttentionBlock_81930796138796
// MI455X (gfx1250) — matrix operand order repaired and checked by reading, not compiled, not run on hardware
//
#include <hip/hip_runtime.h>

#define B_   16
#define N_   3136
#define D_   128
#define Hh_  56
#define Ww_  56
#define NH_  8
#define C2_  256
#define CF_  512
#define EPS_ 1e-5f
#define KC_  256   // K chunk staged in LDS per TDM load

typedef __attribute__((ext_vector_type(16))) __bf16 v16bf;
typedef __attribute__((ext_vector_type(8)))  float  v8f;
typedef __attribute__((ext_vector_type(4)))  unsigned int u32x4;
typedef __attribute__((ext_vector_type(8)))  int i32x8;

// ---------------------------------------------------------------------------
// bf16 WMMA GEMM: C[M,Ncols] = A[M,K] * B[K,Ncols] (+bias)(+resid)
// B is supplied PRE-TRANSPOSED: BT[Ncols][K] row-major.
// Tile 128x64, K chunked at KC_=256: the 64 x kcLen B panel is loaded by one
// TDM tensor_load_to_lds (wave 0) with per-row 4-dword padding; A is staged
// per 32-K step with a register double buffer. 8 waves x (2x2) WMMA tiles.
// ---------------------------------------------------------------------------
__global__ __launch_bounds__(256) void gemm_bf16_wmma(
    const __bf16* __restrict__ A, const __bf16* __restrict__ BT,
    float* __restrict__ Cf, __bf16* __restrict__ Cb,
    const float* __restrict__ bias, const float* __restrict__ resid,
    int M, int Ncols, int K,
    long long sAb, long long sBb, long long sCb)
{
    __shared__ __bf16 As[128][40];        // 128 rows x 32 K (+8 pad)
    __shared__ __bf16 Bsw[64 * (KC_ + 8)]; // B panel, row stride kcLen+8

    const int bz = blockIdx.z;
    A  += (long long)bz * sAb;
    BT += (long long)bz * sBb;
    const long long cOff = (long long)bz * sCb;

    const int tileM = blockIdx.y * 128;
    const int tileN = blockIdx.x * 64;
    const int tid  = threadIdx.x;
    const int wid  = tid >> 5;
    const int lane = tid & 31;
    const int wm = (wid & 3) * 32;
    const int wn = (wid >> 2) * 32;
    const int frag_idx = lane & 15;
    const int half     = lane >> 4;

    // A-chunk decomposition for this thread (2 x 16B chunks per K-step)
    int ar[2], akc[2];
    {
        int c0 = tid, c1 = tid + 256;
        ar[0] = c0 >> 2; akc[0] = (c0 & 3) * 8;
        ar[1] = c1 >> 2; akc[1] = (c1 & 3) * 8;
    }

    v8f acc[2][2] = {};

    for (int k0 = 0; k0 < K; k0 += KC_) {
        const int kcLen = (K - k0 < KC_) ? (K - k0) : KC_;
        const int rowStride = kcLen + 8;

        // ---- TDM: load BT[tileN..tileN+63][k0..k0+kcLen) into Bsw ----
        if (tid < 32) {
            unsigned long long ga =
                (unsigned long long)(uintptr_t)(BT + (long long)tileN * K + k0);
            unsigned ldsA = (unsigned)(uintptr_t)&Bsw[0];
            int rowDw = kcLen >> 1;                  // dwords per row (pow2)
            int padInterval = (31 - __builtin_clz(rowDw)) - 1; // code: 2dw=0..256dw=7
            u32x4 g0;
            g0[0] = 1u;                              // count=1, user descriptor
            g0[1] = ldsA;                            // lds_addr
            g0[2] = (unsigned)ga;                    // global_addr[31:0]
            g0[3] = (unsigned)((ga >> 32) & 0x1FFFFFFu) | (2u << 30); // addr hi | type=2
            i32x8 g1;
            g1[0] = (1 << 16) | (1 << 20) | (padInterval << 22) | (3 << 25);
                   // data_size=2B | pad_enable | pad_interval | pad_amount=4dw
            g1[1] = (kcLen & 0xFFFF) << 16;          // tensor_dim0 lo16
            g1[2] = (kcLen >> 16) | (64 << 16);      // tensor_dim0 hi | tensor_dim1 lo
            g1[3] = (kcLen << 16);                   // tensor_dim1 hi=0 | tile_dim0
            g1[4] = 64;                              // tile_dim1=64, tile_dim2=0
            g1[5] = K;                               // tensor_dim0_stride lo32
            g1[6] = 0;
            g1[7] = 0;
            asm volatile("tensor_load_to_lds %0, %1" :: "s"(g0), "s"(g1) : "memory");
            __builtin_amdgcn_s_wait_tensorcnt(0);
        }
        __syncthreads();

        // ---- inner K loop: A staged per 32, register double-buffered ----
        uint4 areg[2];
        {
            #pragma unroll
            for (int c = 0; c < 2; ++c) {
                int gr = tileM + ar[c];
                if (gr < M) {
                    areg[c] = *(const uint4*)(A + (long long)gr * K + k0 + akc[c]);
                    __builtin_prefetch(A + (long long)gr * K + k0 + akc[c] + KC_, 0, 0);
                } else {
                    areg[c] = make_uint4(0u, 0u, 0u, 0u);
                }
            }
        }
        for (int kk = 0; kk < kcLen; kk += 32) {
            #pragma unroll
            for (int c = 0; c < 2; ++c)
                *(uint4*)&As[ar[c]][akc[c]] = areg[c];
            __syncthreads();

            if (kk + 32 < kcLen) {   // start next A chunk while WMMA runs
                #pragma unroll
                for (int c = 0; c < 2; ++c) {
                    int gr = tileM + ar[c];
                    if (gr < M)
                        areg[c] = *(const uint4*)(A + (long long)gr * K + k0 + kk + 32 + akc[c]);
                    else
                        areg[c] = make_uint4(0u, 0u, 0u, 0u);
                }
            }

            v16bf a[2], b[2];
            #pragma unroll
            for (int i = 0; i < 2; ++i) {
                const __bf16* ap = &As[wm + i * 16 + frag_idx][0];
                #pragma unroll
                for (int t = 0; t < 8; ++t) a[i][t]     = ap[half * 16 + t];
                #pragma unroll
                for (int t = 0; t < 8; ++t) a[i][8 + t] = ap[half * 16 + 8 + t];
                const __bf16* bp = &Bsw[(wn + i * 16 + frag_idx) * rowStride + kk + half * 16];
                #pragma unroll
                for (int t = 0; t < 16; ++t) b[i][t] = bp[t];
            }
            #pragma unroll
            for (int i = 0; i < 2; ++i)
                #pragma unroll
                for (int j = 0; j < 2; ++j)
                    acc[i][j] = __builtin_amdgcn_wmma_f32_16x16x32_bf16(
                        false, a[i], false, b[j], (short)0, acc[i][j], false, false);
            __syncthreads();
        }
    }

    // Epilogue: lane gives n, VGPR r gives M = half*8 + r
    #pragma unroll
    for (int i = 0; i < 2; ++i) {
        #pragma unroll
        for (int j = 0; j < 2; ++j) {
            int col = tileN + wn + j * 16 + frag_idx;
            #pragma unroll
            for (int r = 0; r < 8; ++r) {
                int row = tileM + wm + i * 16 + half * 8 + r;
                if (row < M) {
                    float v = acc[i][j][r];
                    if (bias)  v += bias[col];
                    long long idx = cOff + (long long)row * Ncols + col;
                    if (resid) v += resid[idx];
                    if (Cf) Cf[idx] = v;
                    if (Cb) Cb[idx] = (__bf16)v;
                }
            }
        }
    }
}

// ---------------------------------------------------------------------------
// LayerNorm over last dim C. transposed!=0 -> write outb as [B][C][N_] batched
// transpose (used to produce n11^T as a pre-transposed GEMM B operand).
// ---------------------------------------------------------------------------
__global__ __launch_bounds__(256) void ln_kernel(
    const float* __restrict__ x, const float* __restrict__ g,
    const float* __restrict__ b, float* __restrict__ outf,
    __bf16* __restrict__ outb, int C, int transposed)
{
    __shared__ float red[256];
    long long row = blockIdx.x;
    int tid = threadIdx.x;
    float v = (tid < C) ? x[row * C + tid] : 0.f;
    red[tid] = v;
    __syncthreads();
    for (int s = 128; s > 0; s >>= 1) { if (tid < s) red[tid] += red[tid + s]; __syncthreads(); }
    float mean = red[0] / (float)C;
    __syncthreads();
    float d = (tid < C) ? (v - mean) : 0.f;
    red[tid] = d * d;
    __syncthreads();
    for (int s = 128; s > 0; s >>= 1) { if (tid < s) red[tid] += red[tid + s]; __syncthreads(); }
    float rs = rsqrtf(red[0] / (float)C + EPS_);
    if (tid < C) {
        float o = d * rs * g[tid] + b[tid];
        if (outf) outf[row * C + tid] = o;
        if (outb) {
            if (transposed) {
                long long bb = row / N_, n = row % N_;
                outb[(bb * C + tid) * N_ + n] = (__bf16)o;
            } else {
                outb[row * C + tid] = (__bf16)o;
            }
        }
    }
}

// ---------------------------------------------------------------------------
// Channel attention 16x16 cosine-sim weights per (b,h).
// ---------------------------------------------------------------------------
__global__ __launch_bounds__(256) void ca_stats_kernel(
    const __bf16* __restrict__ qkv, const float* __restrict__ temp,
    float* __restrict__ attnw)
{
    __shared__ float Ss[256], q2s[16], k2s[16];
    int bh = blockIdx.x;
    int b = bh / NH_, h = bh % NH_;
    int tid = threadIdx.x;
    int i = tid >> 4, j = tid & 15;
    float s = 0.f, sq2 = 0.f, sk2 = 0.f;
    long long base = (long long)b * N_ * 384;
    for (int n = 0; n < N_; ++n) {
        long long r = base + (long long)n * 384;
        float q = (float)qkv[r + h * 16 + i];
        float k = (float)qkv[r + 128 + h * 16 + j];
        s += q * k;
        if (j == 0) sq2 += q * q;
        if (i == 0) sk2 += k * k;
    }
    Ss[tid] = s;
    if (j == 0) q2s[i] = sq2;
    if (i == 0) k2s[j] = sk2;
    __syncthreads();
    if (tid < 16) {
        int ii = tid;
        float t = temp[h];
        float nq = fmaxf(sqrtf(q2s[ii]), 1e-12f);
        float row[16], mx = -3.0e38f;
        #pragma unroll
        for (int jj = 0; jj < 16; ++jj) {
            float nk = fmaxf(sqrtf(k2s[jj]), 1e-12f);
            row[jj] = Ss[ii * 16 + jj] / (nq * nk) * t;
            mx = fmaxf(mx, row[jj]);
        }
        float sum = 0.f;
        #pragma unroll
        for (int jj = 0; jj < 16; ++jj) { row[jj] = expf(row[jj] - mx); sum += row[jj]; }
        float inv = 1.f / sum;
        long long o = ((long long)bh * 16 + ii) * 16;
        #pragma unroll
        for (int jj = 0; jj < 16; ++jj) attnw[o + jj] = row[jj] * inv;
    }
}

__global__ __launch_bounds__(128) void ca_apply_kernel(
    const __bf16* __restrict__ qkv, const float* __restrict__ attnw,
    __bf16* __restrict__ co)
{
    long long row = blockIdx.x;
    int b = (int)(row / N_);
    int c = threadIdx.x;
    int h = c >> 4, i = c & 15;
    long long vb = row * 384 + 256 + h * 16;
    long long ab = (((long long)b * NH_ + h) * 16 + i) * 16;
    float s = 0.f;
    #pragma unroll
    for (int j = 0; j < 16; ++j) s += attnw[ab + j] * (float)qkv[vb + j];
    co[row * 128 + c] = (__bf16)s;
}

__global__ __launch_bounds__(128) void key_stats_kernel(
    const float* __restrict__ n22, float* __restrict__ kmax, float* __restrict__ ksum)
{
    int b = blockIdx.x, k = threadIdx.x;
    long long base = (long long)b * N_ * 128 + k;
    float m = -3.0e38f;
    for (int n = 0; n < N_; ++n) m = fmaxf(m, n22[base + (long long)n * 128]);
    float s = 0.f;
    for (int n = 0; n < N_; ++n) s += expf(n22[base + (long long)n * 128] - m);
    kmax[b * 128 + k] = m;
    ksum[b * 128 + k] = s;
}

__global__ __launch_bounds__(256) void exp_key_kernel(
    const float* __restrict__ n22, const float* __restrict__ kmax,
    __bf16* __restrict__ eb)
{
    int bk = blockIdx.x;
    int b = bk / 128, k = bk % 128;
    float m = kmax[bk];
    long long src = (long long)b * N_ * 128 + k;
    long long dst = (long long)bk * N_;
    for (int n = threadIdx.x; n < N_; n += 256)
        eb[dst + n] = (__bf16)expf(n22[src + (long long)n * 128] - m);
}

// ctxbT[b,v,k] = ctxf[b,k,v] / ksum[b,k]   (transposed -> GEMM BT operand)
__global__ __launch_bounds__(256) void ctx_scale_kernel(
    const float* __restrict__ ctxf, const float* __restrict__ ksum,
    __bf16* __restrict__ ctxbT)
{
    long long idx = (long long)blockIdx.x * 256 + threadIdx.x;
    int b = (int)(idx >> 14);
    int v = (int)((idx >> 7) & 127);
    int k = (int)(idx & 127);
    ctxbT[idx] = (__bf16)(ctxf[((long long)b * 128 + k) * 128 + v] / ksum[b * 128 + k]);
}

__global__ __launch_bounds__(128) void query_softmax_kernel(
    const float* __restrict__ n22, __bf16* __restrict__ qb)
{
    __shared__ float red[128];
    long long row = blockIdx.x;
    int tid = threadIdx.x;
    float v = n22[row * 128 + tid];
    red[tid] = v;
    __syncthreads();
    for (int s = 64; s > 0; s >>= 1) { if (tid < s) red[tid] = fmaxf(red[tid], red[tid + s]); __syncthreads(); }
    float mx = red[0];
    __syncthreads();
    float e = expf(v - mx);
    red[tid] = e;
    __syncthreads();
    for (int s = 64; s > 0; s >>= 1) { if (tid < s) red[tid] += red[tid + s]; __syncthreads(); }
    qb[row * 128 + tid] = (__bf16)(e / red[0]);
}

__global__ __launch_bounds__(256) void tx_kernel(
    const float* __restrict__ rep, const float* __restrict__ g,
    const float* __restrict__ b, const float* __restrict__ ch1,
    const float* __restrict__ ch2, float* __restrict__ out)
{
    __shared__ float red[256];
    long long row = blockIdx.x;
    int e = threadIdx.x;
    float v = rep[row * 256 + e];
    red[e] = v;
    __syncthreads();
    for (int s = 128; s > 0; s >>= 1) { if (e < s) red[e] += red[e + s]; __syncthreads(); }
    float mean = red[0] / 256.f;
    __syncthreads();
    float d = v - mean;
    red[e] = d * d;
    __syncthreads();
    for (int s = 128; s > 0; s >>= 1) { if (e < s) red[e] += red[e + s]; __syncthreads(); }
    float rs = rsqrtf(red[0] / 256.f + EPS_);
    float che = (e < 128) ? ch1[row * 128 + e] : ch2[row * 128 + (e - 128)];
    out[row * 256 + e] = che + d * rs * g[e] + b[e];
}

__global__ __launch_bounds__(256) void dwconv_gelu_kernel(
    const __bf16* __restrict__ h1, const float* __restrict__ w,
    const float* __restrict__ bias, __bf16* __restrict__ hg)
{
    long long row = blockIdx.x;
    int b = (int)(row / N_);
    int n = (int)(row % N_);
    int hy = n / Ww_, wx = n % Ww_;
    long long bb = (long long)b * N_;
    #pragma unroll
    for (int it = 0; it < 2; ++it) {
        int c = threadIdx.x + it * 256;
        float acc = bias[c];
        #pragma unroll
        for (int dy = -1; dy <= 1; ++dy) {
            int yy = hy + dy;
            if (yy < 0 || yy >= Hh_) continue;
            #pragma unroll
            for (int dx = -1; dx <= 1; ++dx) {
                int xx = wx + dx;
                if (xx < 0 || xx >= Ww_) continue;
                acc += w[c * 9 + (dy + 1) * 3 + (dx + 1)] *
                       (float)h1[(bb + yy * Ww_ + xx) * CF_ + c];
            }
        }
        float gl = 0.5f * acc * (1.f + erff(acc * 0.70710678118654752f));
        hg[row * CF_ + c] = (__bf16)gl;
    }
}

__global__ __launch_bounds__(256) void convert_kernel(
    const float* __restrict__ src, __bf16* __restrict__ dst, int n)
{
    int i = blockIdx.x * 256 + threadIdx.x;
    if (i < n) dst[i] = (__bf16)src[i];
}

// src [K][Nc] row-major -> dst [Nc][K] row-major (bf16)
__global__ __launch_bounds__(256) void transpose_convert_kernel(
    const float* __restrict__ src, __bf16* __restrict__ dst, int K, int Nc)
{
    int i = blockIdx.x * 256 + threadIdx.x;
    if (i < K * Nc) {
        int n = i / K, k = i % K;
        dst[i] = (__bf16)src[k * Nc + n];
    }
}

// ---------------------------------------------------------------------------
extern "C" void kernel_launch(void* const* d_in, const int* in_sizes, int n_in,
                              void* d_out, int out_size, void* d_ws, size_t ws_size,
                              hipStream_t stream) {
    const float* x1     = (const float*)d_in[0];
    const float* x2     = (const float*)d_in[1];
    const float* n1_g   = (const float*)d_in[2];
    const float* n1_b   = (const float*)d_in[3];
    const float* n3_g   = (const float*)d_in[4];
    const float* n3_b   = (const float*)d_in[5];
    const float* qkv_w  = (const float*)d_in[6];
    const float* ca_temp= (const float*)d_in[7];
    const float* proj_w = (const float*)d_in[8];
    const float* proj_b = (const float*)d_in[9];
    const float* rep_w  = (const float*)d_in[10];
    const float* rep_b  = (const float*)d_in[11];
    const float* cn_g   = (const float*)d_in[12];
    const float* cn_b   = (const float*)d_in[13];
    const float* n2_g   = (const float*)d_in[14];
    const float* n2_b   = (const float*)d_in[15];
    const float* fc1_w  = (const float*)d_in[16];
    const float* fc1_b  = (const float*)d_in[17];
    const float* dw_w   = (const float*)d_in[18];
    const float* dw_b   = (const float*)d_in[19];
    const float* fc2_w  = (const float*)d_in[20];
    const float* fc2_b  = (const float*)d_in[21];
    float* out = (float*)d_out;

    char* ws = (char*)d_ws;
    size_t off = 0;
    auto alloc = [&](size_t bytes) -> char* {
        off = (off + 255) & ~(size_t)255;
        char* p = ws + off;
        off += bytes;
        return p;
    };
    const long long BN = (long long)B_ * N_;   // 50176

    // Persistent
    float*  ch1 = (float*)alloc(BN * D_ * 4);
    float*  ch2 = (float*)alloc(BN * D_ * 4);
    __bf16* wq  = (__bf16*)alloc((size_t)D_ * 3 * D_ * 2);   // [384][128]
    __bf16* wp  = (__bf16*)alloc((size_t)D_ * D_ * 2);       // [128][128]
    __bf16* wr  = (__bf16*)alloc((size_t)D_ * C2_ * 2);      // [256][128] (= rep_w)
    __bf16* w1  = (__bf16*)alloc((size_t)C2_ * CF_ * 2);     // [512][256]
    __bf16* w2  = (__bf16*)alloc((size_t)CF_ * C2_ * 2);     // [256][512]
    const size_t scratch_base = off;

    // Weight prep: all GEMM B operands stored pre-transposed BT[Ncols][K]
    transpose_convert_kernel<<<(49152 + 255) / 256, 256, 0, stream>>>(qkv_w, wq, 128, 384);
    transpose_convert_kernel<<<(16384 + 255) / 256, 256, 0, stream>>>(proj_w, wp, 128, 128);
    convert_kernel<<<(32768 + 255) / 256, 256, 0, stream>>>(rep_w, wr, 32768); // rep_w IS BT
    transpose_convert_kernel<<<(131072 + 255) / 256, 256, 0, stream>>>(fc1_w, w1, 256, 512);
    transpose_convert_kernel<<<(131072 + 255) / 256, 256, 0, stream>>>(fc2_w, w2, 512, 256);

    // ---- Phase A: channel attention per stream ----
    auto run_stream = [&](const float* x, float* ch) {
        off = scratch_base;
        __bf16* xnb   = (__bf16*)alloc(BN * D_ * 2);
        __bf16* qkvb  = (__bf16*)alloc(BN * 3 * D_ * 2);
        float*  attnw = (float*)alloc((size_t)B_ * NH_ * 256 * 4);
        __bf16* cob   = (__bf16*)alloc(BN * D_ * 2);

        ln_kernel<<<(unsigned)BN, 256, 0, stream>>>(x, n1_g, n1_b, nullptr, xnb, D_, 0);
        gemm_bf16_wmma<<<dim3(6, 392, 1), 256, 0, stream>>>(
            xnb, wq, nullptr, qkvb, nullptr, nullptr, (int)BN, 384, 128, 0, 0, 0);
        ca_stats_kernel<<<B_ * NH_, 256, 0, stream>>>(qkvb, ca_temp, attnw);
        ca_apply_kernel<<<(unsigned)BN, 128, 0, stream>>>(qkvb, attnw, cob);
        gemm_bf16_wmma<<<dim3(2, 392, 1), 256, 0, stream>>>(
            cob, wp, ch, nullptr, proj_b, x, (int)BN, 128, 128, 0, 0, 0);
    };
    run_stream(x1, ch1);
    run_stream(x2, ch2);

    // ---- Phase B: linear cross attention ----
    off = scratch_base;
    __bf16* n11t = (__bf16*)alloc(BN * D_ * 2);   // [B][128][N]  (BT for context GEMM)
    float*  n22f = (float*)alloc(BN * D_ * 4);
    __bf16* eb   = (__bf16*)alloc(BN * D_ * 2);   // [B][128][N]  (A for context GEMM)
    float*  kmax = (float*)alloc((size_t)B_ * 128 * 4);
    float*  ksum = (float*)alloc((size_t)B_ * 128 * 4);
    float*  ctxf = (float*)alloc((size_t)B_ * 128 * 128 * 4);
    __bf16* ctxt = (__bf16*)alloc((size_t)B_ * 128 * 128 * 2); // [B][v][k] (BT for att GEMM)
    __bf16* qb   = (__bf16*)alloc(BN * D_ * 2);
    __bf16* attb = (__bf16*)alloc(BN * D_ * 2);
    float*  repf = (float*)alloc(BN * C2_ * 4);

    ln_kernel<<<(unsigned)BN, 256, 0, stream>>>(ch1, n3_g, n3_b, nullptr, n11t, D_, 1);
    ln_kernel<<<(unsigned)BN, 256, 0, stream>>>(ch2, n3_g, n3_b, n22f, nullptr, D_, 0);
    key_stats_kernel<<<B_, 128, 0, stream>>>(n22f, kmax, ksum);
    exp_key_kernel<<<B_ * 128, 256, 0, stream>>>(n22f, kmax, eb);
    // context[b](k,v) = eb[b](128 x N) @ n11[b](N x 128); BT = n11t, K = N_
    gemm_bf16_wmma<<<dim3(2, 1, B_), 256, 0, stream>>>(
        eb, n11t, ctxf, nullptr, nullptr, nullptr, 128, 128, N_,
        (long long)128 * N_, (long long)128 * N_, (long long)128 * 128);
    ctx_scale_kernel<<<(B_ * 128 * 128) / 256, 256, 0, stream>>>(ctxf, ksum, ctxt);
    query_softmax_kernel<<<(unsigned)BN, 128, 0, stream>>>(n22f, qb);
    // att_t[b] = Q[b](N x 128) @ ctx[b](128 x 128); BT = ctxt
    gemm_bf16_wmma<<<dim3(2, 25, B_), 256, 0, stream>>>(
        qb, ctxt, nullptr, attb, nullptr, nullptr, N_, 128, 128,
        (long long)N_ * 128, (long long)128 * 128, (long long)N_ * 128);
    // rep = att_t @ rep_w^T + rep_b  (BT of rep_w^T is rep_w itself)
    gemm_bf16_wmma<<<dim3(4, 392, 1), 256, 0, stream>>>(
        attb, wr, repf, nullptr, rep_b, nullptr, (int)BN, 256, 128, 0, 0, 0);
    tx_kernel<<<(unsigned)BN, 256, 0, stream>>>(repf, cn_g, cn_b, ch1, ch2, out);

    // ---- Phase C: MixFFN ----
    off = scratch_base;
    __bf16* txnb = (__bf16*)alloc(BN * C2_ * 2);
    __bf16* h1b  = (__bf16*)alloc(BN * CF_ * 2);
    __bf16* hgb  = (__bf16*)alloc(BN * CF_ * 2);

    ln_kernel<<<(unsigned)BN, 256, 0, stream>>>(out, n2_g, n2_b, nullptr, txnb, C2_, 0);
    gemm_bf16_wmma<<<dim3(8, 392, 1), 256, 0, stream>>>(
        txnb, w1, nullptr, h1b, fc1_b, nullptr, (int)BN, 512, 256, 0, 0, 0);
    dwconv_gelu_kernel<<<(unsigned)BN, 256, 0, stream>>>(h1b, dw_w, dw_b, hgb);
    gemm_bf16_wmma<<<dim3(4, 392, 1), 256, 0, stream>>>(
        hgb, w2, out, nullptr, fc2_b, out, (int)BN, 256, 512, 0, 0, 0);
}
